// MultiAttentionHead_84765474554570
// MI455X (gfx1250) — matrix operand order repaired and checked by reading, not compiled, not run on hardware
//
#include <hip/hip_runtime.h>
#include <hip/hip_bf16.h>

typedef __attribute__((ext_vector_type(16))) _Float16 v16h;
typedef __attribute__((ext_vector_type(8)))  _Float16 v8h;
typedef __attribute__((ext_vector_type(4)))  _Float16 v4h;
typedef __attribute__((ext_vector_type(2)))  _Float16 v2h;
typedef __attribute__((ext_vector_type(8)))  float    v8f;

constexpr int Bc  = 4;
constexpr int Sc  = 2048;
constexpr int Dc  = 1024;
constexpr int Hc  = 16;
constexpr int HDc = 64;
constexpr int LDSROW = 40;   // padded row stride (halves) to dodge bank conflicts

// Q pre-scale: 1/sqrt(HD) * log2(e)  -> softmax computed in base-2 domain
#define QSCALE 0.18033688011112042f

#if __has_builtin(__builtin_amdgcn_exp2f)
#define EXP2(x) __builtin_amdgcn_exp2f(x)
#else
#define EXP2(x) exp2f(x)
#endif

__device__ __forceinline__ v8f wmma16(v16h a, v16h b, v8f c) {
  return __builtin_amdgcn_wmma_f32_16x16x32_f16(false, a, false, b, (short)0, c, false, false);
}

// A operand (16x32 f16, M x K), row-major source, leading dim ld (halves).
// lane<16: row=lane, K=0..15; lane>=16: row=lane-16, K=16..31 (contiguous 16 halves)
__device__ __forceinline__ v16h load_a(const _Float16* base, int ld, int lane) {
  const int row = lane & 15;
  const int k0  = (lane < 16) ? 0 : 16;
  const _Float16* p = base + (size_t)row * ld + k0;
  v8h lo = *(const v8h*)(p);
  v8h hi = *(const v8h*)(p + 8);
  v16h r;
#pragma unroll
  for (int i = 0; i < 8; ++i) { r[i] = lo[i]; r[i + 8] = hi[i]; }
  return r;
}

// B operand (32x16 f16, K x N) from column-contiguous storage: source row n holds B[:,n].
// lane<16: col=lane, K=0..15; lane>=16: col=lane-16, K=16..31 (contiguous 16 halves)
__device__ __forceinline__ v16h load_bt(const _Float16* base, int ld, int lane) {
  const int col = lane & 15;
  const int k0  = (lane < 16) ? 0 : 16;
  const _Float16* p = base + (size_t)col * ld + k0;
  v8h lo = *(const v8h*)(p);
  v8h hi = *(const v8h*)(p + 8);
  v16h r;
#pragma unroll
  for (int i = 0; i < 8; ++i) { r[i] = lo[i]; r[i + 8] = hi[i]; }
  return r;
}

// ---------------- Kernel 0: weight prep (transpose + fp16 convert) ----------------
__global__ __launch_bounds__(256) void wprep_kernel(
    const float* __restrict__ Wq, const float* __restrict__ Wk,
    const float* __restrict__ Wv, const float* __restrict__ Wo,
    _Float16* __restrict__ WqT, _Float16* __restrict__ WkT,
    _Float16* __restrict__ WvT, _Float16* __restrict__ Wo16) {
  int idx = blockIdx.x * 256 + threadIdx.x;               // over H*D*HD == D*D
  int e = idx % HDc;
  int d = (idx / HDc) % Dc;
  int h = idx / (HDc * Dc);
  size_t t = ((size_t)h * HDc + e) * Dc + d;
  WqT[t] = (_Float16)Wq[idx];
  WkT[t] = (_Float16)Wk[idx];
  WvT[t] = (_Float16)Wv[idx];
  Wo16[idx] = (_Float16)Wo[idx];
}

// ---------------- Kernel 1: LayerNorm, f32 -> f16 ----------------
__global__ __launch_bounds__(256) void ln_kernel(
    const float* __restrict__ x, const float* __restrict__ gamma,
    const float* __restrict__ beta, _Float16* __restrict__ xn) {
  const int lane = threadIdx.x & 31;
  const int wave = threadIdx.x >> 5;
  const int row  = blockIdx.x * 8 + wave;   // 0 .. B*S-1
  const float* xr = x + (size_t)row * Dc;
  float vals[32];
  float sum = 0.f, sumsq = 0.f;
#pragma unroll
  for (int i = 0; i < 8; ++i) {
    float4 f = *(const float4*)(xr + lane * 4 + i * 128);
    vals[i * 4 + 0] = f.x; vals[i * 4 + 1] = f.y;
    vals[i * 4 + 2] = f.z; vals[i * 4 + 3] = f.w;
    sum   += f.x + f.y + f.z + f.w;
    sumsq += f.x * f.x + f.y * f.y + f.z * f.z + f.w * f.w;
  }
#pragma unroll
  for (int m = 1; m < 32; m <<= 1) {
    sum   += __shfl_xor(sum,   m, 32);
    sumsq += __shfl_xor(sumsq, m, 32);
  }
  const float mu = sum * (1.0f / Dc);
  const float var = sumsq * (1.0f / Dc) - mu * mu;
  const float rs = rsqrtf(var + 1e-5f);
  _Float16* xo = xn + (size_t)row * Dc;
#pragma unroll
  for (int i = 0; i < 8; ++i) {
    int idx = lane * 4 + i * 128;
    v4h o;
#pragma unroll
    for (int j = 0; j < 4; ++j)
      o[j] = (_Float16)((vals[i * 4 + j] - mu) * rs * gamma[idx + j] + beta[idx + j]);
    *(v4h*)(xo + idx) = o;
  }
}

// ---------------- Kernel 2: QKV projection GEMMs ----------------
// Each wave: 16 rows x full HD(=64) for Q,K,V of one (b,h).
// Q pre-scaled by 1/sqrt(HD)*log2(e). Q,K stored [b][h][S][HD]; V stored [b][h][HD][S].
__global__ __launch_bounds__(256) void qkv_kernel(
    const _Float16* __restrict__ xn,
    const _Float16* __restrict__ WqT, const _Float16* __restrict__ WkT,
    const _Float16* __restrict__ WvT,
    _Float16* __restrict__ Q, _Float16* __restrict__ Km, _Float16* __restrict__ Vt) {
  const int lane = threadIdx.x & 31;
  const int wave = threadIdx.x >> 5;
  const int m0 = blockIdx.x * 128 + wave * 16;
  const int h = blockIdx.y, b = blockIdx.z;
  const _Float16* xa = xn + ((size_t)b * Sc + m0) * Dc;
  const _Float16* wq = WqT + (size_t)h * HDc * Dc;
  const _Float16* wk = WkT + (size_t)h * HDc * Dc;
  const _Float16* wv = WvT + (size_t)h * HDc * Dc;
  v8f aq[4] = {}, ak[4] = {}, av[4] = {};
  for (int k0 = 0; k0 < Dc; k0 += 32) {
    v16h a = load_a(xa + k0, Dc, lane);
#pragma unroll
    for (int n = 0; n < 4; ++n) {
      size_t off = (size_t)(n * 16) * Dc + k0;
      aq[n] = wmma16(a, load_bt(wq + off, Dc, lane), aq[n]);
      ak[n] = wmma16(a, load_bt(wk + off, Dc, lane), ak[n]);
      av[n] = wmma16(a, load_bt(wv + off, Dc, lane), av[n]);
    }
  }
  const int col = lane & 15;
  const int rowoff = (lane < 16) ? 0 : 8;
  const size_t bh = (size_t)(b * Hc + h);
  _Float16* qp = Q + (bh * Sc + m0) * HDc;
  _Float16* kp = Km + (bh * Sc + m0) * HDc;
  _Float16* vp = Vt + bh * HDc * Sc;
#pragma unroll
  for (int n = 0; n < 4; ++n) {
    const int e = n * 16 + col;
#pragma unroll
    for (int r = 0; r < 8; ++r) {
      const int row = rowoff + r;
      qp[(size_t)row * HDc + e] = (_Float16)(aq[n][r] * QSCALE);
      kp[(size_t)row * HDc + e] = (_Float16)ak[n][r];
      vp[(size_t)e * Sc + (m0 + row)] = (_Float16)av[n][r];
    }
  }
}

// ---------------- Kernel 3: causal flash attention (transposed scores) ----------------
// S^T = K * Q^T (key = M in-register, query = N lane); softmax in base-2 domain.
// Interior key tiles skip causal masking entirely; only diagonal tiles mask.
template <bool MASK>
__device__ __forceinline__ void attn_tile(
    int key0, int qrow, int hoff, int col, int lane,
    const _Float16* __restrict__ kbase, const _Float16* __restrict__ vbase,
    _Float16* pst, const v16h& bq0, const v16h& bq1,
    float& m, float& l, v8f (&o)[4]) {
  const _Float16* kp = kbase + (size_t)key0 * HDc;
  v8f st0 = {}, st1 = {};
  st0 = wmma16(load_a(kp, HDc, lane), bq0, st0);
  st0 = wmma16(load_a(kp + 32, HDc, lane), bq1, st0);
  st1 = wmma16(load_a(kp + 16 * HDc, HDc, lane), bq0, st1);
  st1 = wmma16(load_a(kp + 16 * HDc + 32, HDc, lane), bq1, st1);

  float v0[8], v1[8];
  float mloc = -1e30f;
#pragma unroll
  for (int r = 0; r < 8; ++r) {
    if (MASK) {
      const int ki = key0 + hoff + r;
      v0[r] = (ki <= qrow) ? st0[r] : -1e30f;
      v1[r] = (ki + 16 <= qrow) ? st1[r] : -1e30f;
    } else {
      v0[r] = st0[r];
      v1[r] = st1[r];
    }
    mloc = fmaxf(mloc, fmaxf(v0[r], v1[r]));
  }
  mloc = fmaxf(mloc, __shfl_xor(mloc, 16, 32));
  const float mnew = fmaxf(m, mloc);
  const float alpha = EXP2(m - mnew);
  m = mnew;

  float p0[8], p1[8];
  float sum = 0.f;
#pragma unroll
  for (int r = 0; r < 8; ++r) {
    p0[r] = EXP2(v0[r] - mnew);
    p1[r] = EXP2(v1[r] - mnew);
    sum += p0[r] + p1[r];
  }
  sum += __shfl_xor(sum, 16, 32);
  l = l * alpha + sum;

  // stage P^T into LDS as [query][key] (keys contiguous), packed b32 stores
  {
    _Float16* wp = pst + col * LDSROW + hoff;
#pragma unroll
    for (int r2 = 0; r2 < 4; ++r2) {
      v2h t0; t0[0] = (_Float16)p0[2 * r2]; t0[1] = (_Float16)p0[2 * r2 + 1];
      *(v2h*)(wp + 2 * r2) = t0;
      v2h t1; t1[0] = (_Float16)p1[2 * r2]; t1[1] = (_Float16)p1[2 * r2 + 1];
      *(v2h*)(wp + 16 + 2 * r2) = t1;
    }
  }
  asm volatile("s_wait_dscnt 0" ::: "memory");

  // P^T B operand (32 keys x 16 queries)
  v16h pb;
  {
    const int kk = (lane < 16) ? 0 : 16;
    const _Float16* pp = pst + col * LDSROW + kk;
    v8h lo = *(const v8h*)(pp);
    v8h hi = *(const v8h*)(pp + 8);
#pragma unroll
    for (int i = 0; i < 8; ++i) { pb[i] = lo[i]; pb[i + 8] = hi[i]; }
  }

#pragma unroll
  for (int n = 0; n < 4; ++n)
#pragma unroll
    for (int r = 0; r < 8; ++r) o[n][r] *= alpha;

  const _Float16* vp = vbase + key0;
#pragma unroll
  for (int n = 0; n < 4; ++n)
    o[n] = wmma16(load_a(vp + (size_t)(n * 16) * Sc, Sc, lane), pb, o[n]);
}

__global__ __launch_bounds__(256) void attn_kernel(
    const _Float16* __restrict__ Q, const _Float16* __restrict__ Km,
    const _Float16* __restrict__ Vt, _Float16* __restrict__ Z) {
  __shared__ __align__(16) _Float16 lds[8 * 16 * LDSROW];
  const int lane = threadIdx.x & 31;
  const int wave = threadIdx.x >> 5;
  const int q0 = (blockIdx.x * 8 + wave) * 16;
  const int h = blockIdx.y, b = blockIdx.z;
  _Float16* pst = lds + wave * 16 * LDSROW;
  const size_t bh = (size_t)(b * Hc + h);
  const _Float16* qp = Q + (bh * Sc + q0) * HDc;
  const v16h bq0 = load_bt(qp, HDc, lane);
  const v16h bq1 = load_bt(qp + 32, HDc, lane);
  const int col  = lane & 15;           // query within tile
  const int hoff = (lane < 16) ? 0 : 8; // M-offset of this lane's C rows
  const int qrow = q0 + col;            // absolute query index
  const _Float16* kbase = Km + bh * Sc * HDc;
  const _Float16* vbase = Vt + bh * HDc * Sc;

  float m = -1e30f, l = 0.f;
  v8f o[4] = {};                        // O^T: hd x 16 queries (4 M-tiles of 16)

  const int jn    = (q0 + 47) / 32;     // total 32-key tiles
  const int jfull = (q0 + 1) / 32;      // tiles fully below the diagonal (no masking)
  for (int j = 0; j < jfull; ++j) {
    const int key0 = j * 32;
    __builtin_prefetch(kbase + (size_t)(key0 + 32) * HDc + lane * 64, 0, 1);
    __builtin_prefetch(vbase + (key0 + 32) + (size_t)(lane & 3) * 16 * Sc + (lane >> 2) * 8 * Sc, 0, 1);
    attn_tile<false>(key0, qrow, hoff, col, lane, kbase, vbase, pst, bq0, bq1, m, l, o);
  }
  for (int j = jfull; j < jn; ++j)
    attn_tile<true>(j * 32, qrow, hoff, col, lane, kbase, vbase, pst, bq0, bq1, m, l, o);

  // epilogue: Z[b][q0+col][h*64 + n*16 + hoff + r], contiguous v8h per tile
  const float invl = 1.f / l;
  _Float16* zp = Z + ((size_t)b * Sc + (q0 + col)) * Dc + h * HDc;
#pragma unroll
  for (int n = 0; n < 4; ++n) {
    v8h zo;
#pragma unroll
    for (int r = 0; r < 8; ++r) zo[r] = (_Float16)(o[n][r] * invl);
    *(v8h*)(zp + n * 16 + hoff) = zo;
  }
}

// ---------------- Kernel 4: output projection + bias ----------------
__global__ __launch_bounds__(256) void oproj_kernel(
    const _Float16* __restrict__ Zm, const _Float16* __restrict__ Wo16,
    const float* __restrict__ bo, float* __restrict__ out) {
  const int lane = threadIdx.x & 31;
  const int wave = threadIdx.x >> 5;
  const int m0 = blockIdx.x * 128 + wave * 16;
  const int n0 = blockIdx.y * 64;
  const _Float16* za = Zm + (size_t)m0 * Dc;
  v8f acc[4] = {};
  for (int k0 = 0; k0 < Dc; k0 += 32) {
    v16h a = load_a(za + k0, Dc, lane);
#pragma unroll
    for (int n = 0; n < 4; ++n)
      acc[n] = wmma16(a, load_bt(Wo16 + (size_t)(n0 + n * 16) * Dc + k0, Dc, lane), acc[n]);
  }
  const int col = lane & 15;
  const int rowoff = (lane < 16) ? 0 : 8;
#pragma unroll
  for (int n = 0; n < 4; ++n) {
    const float bias = bo[n0 + n * 16 + col];
#pragma unroll
    for (int r = 0; r < 8; ++r)
      out[(size_t)(m0 + rowoff + r) * Dc + n0 + n * 16 + col] = acc[n][r] + bias;
  }
}

extern "C" void kernel_launch(void* const* d_in, const int* in_sizes, int n_in,
                              void* d_out, int out_size, void* d_ws, size_t ws_size,
                              hipStream_t stream) {
  const float* x     = (const float*)d_in[0];
  const float* Wq    = (const float*)d_in[1];
  const float* Wk    = (const float*)d_in[2];
  const float* Wv    = (const float*)d_in[3];
  const float* Wo    = (const float*)d_in[4];
  const float* bo    = (const float*)d_in[5];
  const float* gamma = (const float*)d_in[6];
  const float* beta  = (const float*)d_in[7];
  float* out = (float*)d_out;

  char* ws = (char*)d_ws;
  const size_t szBSD = (size_t)Bc * Sc * Dc * sizeof(_Float16);  // 16 MB
  const size_t szW   = (size_t)Hc * HDc * Dc * sizeof(_Float16); //  2 MB
  _Float16* xn  = (_Float16*)ws;            ws += szBSD;
  _Float16* Qb  = (_Float16*)ws;            ws += szBSD;
  _Float16* Kb  = (_Float16*)ws;            ws += szBSD;
  _Float16* Vtb = (_Float16*)ws;            ws += szBSD;
  _Float16* Zb  = (_Float16*)ws;            ws += szBSD;
  _Float16* WqT = (_Float16*)ws;            ws += szW;
  _Float16* WkT = (_Float16*)ws;            ws += szW;
  _Float16* WvT = (_Float16*)ws;            ws += szW;
  _Float16* Wo16 = (_Float16*)ws;           ws += szW;

  wprep_kernel<<<(Hc * Dc * HDc) / 256, 256, 0, stream>>>(Wq, Wk, Wv, Wo, WqT, WkT, WvT, Wo16);
  ln_kernel<<<(Bc * Sc) / 8, 256, 0, stream>>>(x, gamma, beta, xn);
  qkv_kernel<<<dim3(Sc / 128, Hc, Bc), 256, 0, stream>>>(xn, WqT, WkT, WvT, Qb, Kb, Vtb);
  attn_kernel<<<dim3(Sc / 128, Hc, Bc), 256, 0, stream>>>(Qb, Kb, Vtb, Zb);
  oproj_kernel<<<dim3((Bc * Sc) / 128, Dc / 64), 256, 0, stream>>>(Zb, Wo16, bo, out);
}
